// CGWeight_52407190946312
// MI455X (gfx1250) — matrix operand order repaired and checked by reading, not compiled, not run on hardware
//
#include <hip/hip_runtime.h>

typedef _Float16 half16 __attribute__((ext_vector_type(16)));
typedef _Float16 half2v __attribute__((ext_vector_type(2)));
typedef float    float8 __attribute__((ext_vector_type(8)));
typedef unsigned int uint4v __attribute__((ext_vector_type(4)));

// ---------------------------------------------------------------------------
// Fused constant build: M[n][j1][j2] = w[combo(la,lh), n/5] * cg[r1,r2,n%5]
// packed to f16 in the exact 32x16 B-operand VGPR layout of
// v_wmma_f32_16x16x32_f16 (lane = g*16+n ; vgpr v holds rows K=16(v>>2)+8g+2(v&3), +1).
// kk = j2*16 + j1 over the concatenated 16-dim a / h irreps.
// ---------------------------------------------------------------------------
struct CgArgs {
  const float* cg[11];
  const float* w;
  unsigned int* bws;   // 2 ntiles x 8 chunks x 32 lanes x 8 vgprs = 4096 dwords
};

__device__ __forceinline__ float fused_M(const CgArgs& A, int n, int kk) {
  if (n >= 20) return 0.0f;
  const int blk[16] = {0,1,1,1,2,2,2,2,2,3,3,3,3,3,3,3};
  const int off[4]  = {0,1,4,9};
  const int dim[4]  = {1,3,5,7};
  const int cidx[4][4] = {{-1,-1,0,-1},{-1,1,2,3},{4,5,6,7},{-1,8,9,10}};
  int j1 = kk & 15, j2 = kk >> 4;
  int la = blk[j1], lh = blk[j2];
  int i = cidx[la][lh];
  if (i < 0) return 0.0f;                       // triangle-violating block: zero
  int r1 = j1 - off[la], r2 = j2 - off[lh];
  float cg = A.cg[i][(r1 * dim[lh] + r2) * 5 + (n % 5)];
  return A.w[i * 4 + (n / 5)] * cg;
}

__global__ void cg_prep_kernel(CgArgs args) {
  int t = blockIdx.x * blockDim.x + threadIdx.x;
  if (t >= 4096) return;
  int v = t & 7, lane = (t >> 3) & 31, c = (t >> 8) & 7, nt = t >> 11;
  int g = lane >> 4, nl = lane & 15;
  int n  = nt * 16 + nl;
  int k0 = 32 * c + 16 * (v >> 2) + 8 * g + 2 * (v & 3); // global kk of half0 (table order); half1 = k0+1
  union { _Float16 h[2]; unsigned int u; } pk;
  pk.h[0] = (_Float16)fused_M(args, n, k0);
  pk.h[1] = (_Float16)fused_M(args, n, k0 + 1);
  args.bws[((nt * 8 + c) * 32 + lane) * 8 + v] = pk.u;
}

// ---------------------------------------------------------------------------
// Main kernel: one wave owns 16 edges per iteration.
// D(16 edges x 16 outs) += A(16x32 products) x B(32x16 fused CG), K = 256.
// ---------------------------------------------------------------------------
__global__ __launch_bounds__(256) void cg_wmma_kernel(
    const float* __restrict__ a0, const float* __restrict__ a1,
    const float* __restrict__ a2, const float* __restrict__ a3,
    const float* __restrict__ h0p, const float* __restrict__ h1p,
    const float* __restrict__ h2p, const float* __restrict__ h3p,
    const unsigned int* __restrict__ bws, float* __restrict__ out,
    int ntiles, int E) {
  __shared__ __align__(16) unsigned int Bsh[4096];
  for (int i = threadIdx.x; i < 4096; i += 256) Bsh[i] = bws[i];
  __syncthreads();

  const int lane = threadIdx.x & 31;
  const int wave = threadIdx.x >> 5;
  const int g = lane >> 4;        // K-half group
  const int m = lane & 15;        // edge row in A / output column in D
  const int nwaves = gridDim.x * 8;

  for (int tile = blockIdx.x * 8 + wave; tile < ntiles; tile += nwaves) {
    int e = tile * 16 + m;
    if (e >= E) e = E - 1;        // tail clamp (E = 2M is a multiple of 16)
    size_t e3 = (size_t)e * 3, e5 = (size_t)e * 5, e7 = (size_t)e * 7;

    // a-slice j1 in [8g, 8g+8): concat layout a0|a1|a2|a3 (widths 1,3,5,7)
    float af[8];
    if (g == 0) {
      af[0] = a0[e];
      af[1] = a1[e3];   af[2] = a1[e3+1]; af[3] = a1[e3+2];
      af[4] = a2[e5];   af[5] = a2[e5+1]; af[6] = a2[e5+2]; af[7] = a2[e5+3];
    } else {
      af[0] = a2[e5+4];
      af[1] = a3[e7];   af[2] = a3[e7+1]; af[3] = a3[e7+2]; af[4] = a3[e7+3];
      af[5] = a3[e7+4]; af[6] = a3[e7+5]; af[7] = a3[e7+6];
    }
    half2v ap[4];
#pragma unroll
    for (int v = 0; v < 4; ++v) {
      ap[v][0] = (_Float16)af[2*v];
      ap[v][1] = (_Float16)af[2*v+1];
    }

    // full 16-dim h vector for this edge
    _Float16 hh[16];
    hh[0] = (_Float16)h0p[e];
    hh[1] = (_Float16)h1p[e3];   hh[2] = (_Float16)h1p[e3+1]; hh[3] = (_Float16)h1p[e3+2];
    hh[4] = (_Float16)h2p[e5];   hh[5] = (_Float16)h2p[e5+1]; hh[6] = (_Float16)h2p[e5+2];
    hh[7] = (_Float16)h2p[e5+3]; hh[8] = (_Float16)h2p[e5+4];
#pragma unroll
    for (int q = 0; q < 7; ++q) hh[9+q] = (_Float16)h3p[e7+q];

    float8 acc0 = {0,0,0,0,0,0,0,0};
    float8 acc1 = {0,0,0,0,0,0,0,0};

#pragma unroll
    for (int c = 0; c < 8; ++c) {               // K-chunk: j2 = 2c, 2c+1
      half2v hb0; hb0[0] = hh[2*c];   hb0[1] = hh[2*c];
      half2v hb1; hb1[0] = hh[2*c+1]; hb1[1] = hh[2*c+1];
      union { half2v p[8]; half16 v; } A;
#pragma unroll
      for (int v = 0; v < 4; ++v) {             // 8 v_pk_mul_f16 builds A
        A.p[v]     = ap[v] * hb0;
        A.p[v + 4] = ap[v] * hb1;
      }
      union { uint4v q[2]; half16 v; } B0, B1;
      const unsigned int* b0 = &Bsh[(c * 32 + lane) * 8];
      const unsigned int* b1 = &Bsh[((8 + c) * 32 + lane) * 8];
      B0.q[0] = *(const uint4v*)b0; B0.q[1] = *(const uint4v*)(b0 + 4);
      B1.q[0] = *(const uint4v*)b1; B1.q[1] = *(const uint4v*)(b1 + 4);

      acc0 = __builtin_amdgcn_wmma_f32_16x16x32_f16(false, A.v, false, B0.v,
                                                    (short)0, acc0, false, false);
      acc1 = __builtin_amdgcn_wmma_f32_16x16x32_f16(false, A.v, false, B1.v,
                                                    (short)0, acc1, false, false);
    }

    // D layout: lane g*16+n, vgpr r -> edge 8g+r, output column n
    const int ebase = tile * 16 + 8 * g;
#pragma unroll
    for (int r = 0; r < 8; ++r) {
      int eo = ebase + r;
      if (eo < E) {
        out[(size_t)eo * 20 + m] = acc0[r];
        if (m < 4) out[(size_t)eo * 20 + 16 + m] = acc1[r];
      }
    }

    // prefetch the two widest rows of the next tile into cache
    int tn = tile + nwaves;
    if (tn < ntiles) {
      int en = tn * 16 + m;
      if (en < E) {
        __builtin_prefetch(&a3[(size_t)en * 7], 0, 1);
        __builtin_prefetch(&h3p[(size_t)en * 7], 0, 1);
      }
    }
  }
}

// ---------------------------------------------------------------------------
extern "C" void kernel_launch(void* const* d_in, const int* in_sizes, int n_in,
                              void* d_out, int out_size, void* d_ws, size_t ws_size,
                              hipStream_t stream) {
  (void)out_size; (void)ws_size;
  const float* a0 = (const float*)d_in[0];
  const float* a1 = (const float*)d_in[1];
  const float* a2 = (const float*)d_in[2];
  const float* a3 = (const float*)d_in[3];
  const float* h0 = (const float*)d_in[4];
  const float* h1 = (const float*)d_in[5];
  const float* h2 = (const float*)d_in[6];
  const float* h3 = (const float*)d_in[7];

  CgArgs args;
  args.w   = (const float*)d_in[8];
  args.bws = (unsigned int*)d_ws;
  // flat offsets of the 11 CG tensors if they arrive concatenated
  static const int cgoff[11] = {0,25,70,145,250,275,350,475,650,755,930};
  if (n_in >= 20) {
    for (int i = 0; i < 11; ++i) args.cg[i] = (const float*)d_in[9 + i];
  } else {
    const float* base = (const float*)d_in[9];
    for (int i = 0; i < 11; ++i) args.cg[i] = base + cgoff[i];
  }

  const int E = in_sizes[0];            // a0 is (E,1)
  const int ntiles = (E + 15) / 16;

  cg_prep_kernel<<<16, 256, 0, stream>>>(args);

  int blocks = (ntiles + 7) / 8;
  if (blocks > 1024) blocks = 1024;     // ~8K waves, grid-stride over 125K tiles
  cg_wmma_kernel<<<blocks, 256, 0, stream>>>(a0, a1, a2, a3, h0, h1, h2, h3,
                                             (const unsigned int*)d_ws,
                                             (float*)d_out, ntiles, E);
}
